// PointConvD_1692217115334
// MI455X (gfx1250) — matrix operand order repaired and checked by reading, not compiled, not run on hardware
//
#include <hip/hip_runtime.h>
#include <hip/hip_bf16.h>

// Problem constants (match reference)
#define Bv      8
#define Nv      8192
#define Dv      256
#define NPOINTv 2048
#define NSAMPLEv 32
#define WNETv   16
#define IN_CHv  259            // D + 3
#define OUT_CHv 512
#define KDIMv   (IN_CHv * WNETv)   // 4144
#define KPv     4160               // K padded to multiple of 32
#define Mrows   (Bv * NPOINTv)     // 16384
#define KCv     416                // GEMM K-chunk staged in LDS (10 chunks)
#define NCHUNK  (KPv / KCv)        // 10

typedef __attribute__((ext_vector_type(16))) __bf16 v16bf;
typedef __attribute__((ext_vector_type(8)))  float  v8f;
typedef __attribute__((ext_vector_type(4)))  unsigned int v4u;
typedef __attribute__((ext_vector_type(4)))  int v4i;
typedef __attribute__((ext_vector_type(8)))  int v8i;

union ABu { uint4 u[2]; v16bf v; };

#if __has_builtin(__builtin_amdgcn_tensor_load_to_lds)
#define HAVE_TDM 1
#else
#define HAVE_TDM 0
#endif

// ---------------------------------------------------------------------------
// 1) Furthest point sampling: one block per batch, 1024 threads (32 waves).
// ---------------------------------------------------------------------------
__global__ void __launch_bounds__(1024)
fps_kernel(const float* __restrict__ xyz, int* __restrict__ fpsidx) {
  const int b   = blockIdx.x;
  const int tid = threadIdx.x;
  const float* X = xyz + (size_t)(b * 3 + 0) * Nv;
  const float* Y = xyz + (size_t)(b * 3 + 1) * Nv;
  const float* Z = xyz + (size_t)(b * 3 + 2) * Nv;

  float px[8], py[8], pz[8], dd[8];
#pragma unroll
  for (int j = 0; j < 8; ++j) {
    int i = tid + j * 1024;
    px[j] = X[i]; py[j] = Y[i]; pz[j] = Z[i];
    dd[j] = 1e10f;
  }

  __shared__ float swv[32];
  __shared__ int   swi[32];
  __shared__ float cen[3];
  __shared__ int   slast;
  if (tid == 0) { slast = 0; fpsidx[b * NPOINTv + 0] = 0; }
  __syncthreads();

  const int lane = tid & 31, wave = tid >> 5;
  for (int s = 1; s < NPOINTv; ++s) {
    if (tid == 0) { int l = slast; cen[0] = X[l]; cen[1] = Y[l]; cen[2] = Z[l]; }
    __syncthreads();
    const float cx = cen[0], cy = cen[1], cz = cen[2];
    float bv = -1.0f; int bi = 0;
#pragma unroll
    for (int j = 0; j < 8; ++j) {
      float dx = px[j] - cx, dy = py[j] - cy, dz = pz[j] - cz;
      float d = dx * dx + dy * dy + dz * dz;
      dd[j] = fminf(dd[j], d);
      int i = tid + j * 1024;
      if (dd[j] > bv || (dd[j] == bv && i < bi)) { bv = dd[j]; bi = i; }
    }
#pragma unroll
    for (int off = 16; off; off >>= 1) {
      float ov = __shfl_down(bv, off, 32);
      int   oi = __shfl_down(bi, off, 32);
      if (ov > bv || (ov == bv && oi < bi)) { bv = ov; bi = oi; }
    }
    if (lane == 0) { swv[wave] = bv; swi[wave] = bi; }
    __syncthreads();
    if (wave == 0) {
      bv = swv[lane]; bi = swi[lane];
#pragma unroll
      for (int off = 16; off; off >>= 1) {
        float ov = __shfl_down(bv, off, 32);
        int   oi = __shfl_down(bi, off, 32);
        if (ov > bv || (ov == bv && oi < bi)) { bv = ov; bi = oi; }
      }
      if (lane == 0) { fpsidx[b * NPOINTv + s] = bi; slast = bi; }
    }
    __syncthreads();
  }
}

// ---------------------------------------------------------------------------
// 2) Gather new_xyz -> ws [B,S,3] and d_out region 0 [B,3,S]
// ---------------------------------------------------------------------------
__global__ void gather_newxyz_kernel(const float* __restrict__ xyz,
                                     const int* __restrict__ fpsidx,
                                     float* __restrict__ nxz_ws,
                                     float* __restrict__ out0) {
  int t = blockIdx.x * blockDim.x + threadIdx.x;
  if (t >= Bv * NPOINTv) return;
  int b = t >> 11, s = t & (NPOINTv - 1);
  int gi = fpsidx[t];
  const float* X = xyz + (size_t)(b * 3) * Nv;
  float x = X[gi], y = X[Nv + gi], z = X[2 * Nv + gi];
  nxz_ws[t * 3 + 0] = x; nxz_ws[t * 3 + 1] = y; nxz_ws[t * 3 + 2] = z;
  out0[((size_t)b * 3 + 0) * NPOINTv + s] = x;
  out0[((size_t)b * 3 + 1) * NPOINTv + s] = y;
  out0[((size_t)b * 3 + 2) * NPOINTv + s] = z;
}

// ---------------------------------------------------------------------------
// 3) kNN: one wave per query, 8192 distances staged in 32 KB LDS.
// ---------------------------------------------------------------------------
__global__ void __launch_bounds__(32)
knn_kernel(const float* __restrict__ xyz, const float* __restrict__ nxz,
           int* __restrict__ knnidx) {
  const int q = blockIdx.x;
  const int b = q >> 11;
  const int lane = threadIdx.x;
  __shared__ float sd[Nv];
  const float* X = xyz + (size_t)(b * 3) * Nv;
  const float* Y = X + Nv;
  const float* Z = Y + Nv;
  const float qx = nxz[q * 3 + 0], qy = nxz[q * 3 + 1], qz = nxz[q * 3 + 2];
  for (int i = lane; i < Nv; i += 32) {
    float dx = X[i] - qx, dy = Y[i] - qy, dz = Z[i] - qz;
    sd[i] = dx * dx + dy * dy + dz * dz;
  }
  __syncthreads();
  float lv = 3.0e38f; int li = lane;
  for (int i = lane; i < Nv; i += 32) {
    float v = sd[i];
    if (v < lv) { lv = v; li = i; }
  }
  for (int r = 0; r < NSAMPLEv; ++r) {
    float bv = lv; int bi = li;
#pragma unroll
    for (int off = 16; off; off >>= 1) {
      float ov = __shfl_xor(bv, off, 32);
      int   oi = __shfl_xor(bi, off, 32);
      if (ov < bv || (ov == bv && oi < bi)) { bv = ov; bi = oi; }
    }
    if (lane == (bi & 31)) {
      sd[bi] = 3.0e38f;
      lv = 3.0e38f; li = lane;
      for (int i = lane; i < Nv; i += 32) {
        float v = sd[i];
        if (v < lv) { lv = v; li = i; }
      }
    }
    if (lane == 0) knnidx[(size_t)q * NSAMPLEv + r] = bi;
  }
}

// ---------------------------------------------------------------------------
// 4) WeightNet MLP + weighted aggregation -> bf16 GEMM A rows (K-padded).
// ---------------------------------------------------------------------------
__global__ void __launch_bounds__(256)
agg_kernel(const float* __restrict__ xyz, const float* __restrict__ pts,
           const float* __restrict__ w1, const float* __restrict__ b1,
           const float* __restrict__ w2, const float* __restrict__ b2,
           const float* __restrict__ w3, const float* __restrict__ b3,
           const float* __restrict__ nxz, const int* __restrict__ knnidx,
           __hip_bfloat16* __restrict__ Abf) {
  const int q = blockIdx.x;          // GEMM row m = b*S + s
  const int b = q >> 11;
  const int tid = threadIdx.x;
  __shared__ float wmat[NSAMPLEv][WNETv];
  __shared__ float xyzn[NSAMPLEv][3];
  __shared__ int   sidx[NSAMPLEv];

  if (tid < NSAMPLEv) {
    const int k = tid;
    const int gi = knnidx[(size_t)q * NSAMPLEv + k];
    const float* X = xyz + (size_t)(b * 3) * Nv;
    const float nx = X[gi]          - nxz[q * 3 + 0];
    const float ny = X[Nv + gi]     - nxz[q * 3 + 1];
    const float nz = X[2 * Nv + gi] - nxz[q * 3 + 2];
    sidx[k] = gi; xyzn[k][0] = nx; xyzn[k][1] = ny; xyzn[k][2] = nz;
    float h1[8], h2[8];
#pragma unroll
    for (int o = 0; o < 8; ++o) {
      float t = w1[o * 3 + 0] * nx + w1[o * 3 + 1] * ny + w1[o * 3 + 2] * nz + b1[o];
      h1[o] = fmaxf(t, 0.0f);
    }
#pragma unroll
    for (int o = 0; o < 8; ++o) {
      float t = b2[o];
#pragma unroll
      for (int i2 = 0; i2 < 8; ++i2) t += w2[o * 8 + i2] * h1[i2];
      h2[o] = fmaxf(t, 0.0f);
    }
#pragma unroll
    for (int o = 0; o < WNETv; ++o) {
      float t = b3[o];
#pragma unroll
      for (int i2 = 0; i2 < 8; ++i2) t += w3[o * 8 + i2] * h2[i2];
      wmat[k][o] = fmaxf(t, 0.0f);
    }
  }
  __syncthreads();

  for (int c = tid; c < IN_CHv; c += 256) {
    float np[NSAMPLEv];
    if (c < 3) {
#pragma unroll
      for (int k = 0; k < NSAMPLEv; ++k) np[k] = xyzn[k][c];
    } else {
      const float* pc = pts + ((size_t)b * Dv + (c - 3)) * Nv;
#pragma unroll
      for (int k = 0; k < NSAMPLEv; ++k) np[k] = pc[sidx[k]];
    }
    const size_t base = (size_t)q * KPv + (size_t)c * WNETv;
#pragma unroll
    for (int w = 0; w < WNETv; ++w) {
      float sum = 0.0f;
#pragma unroll
      for (int k = 0; k < NSAMPLEv; ++k) sum += np[k] * wmat[k][w];
      Abf[base + w] = __float2bfloat16(sum);
    }
  }
  if (tid < (KPv - KDIMv))   // zero the K padding
    Abf[(size_t)q * KPv + KDIMv + tid] = __float2bfloat16(0.0f);
}

// ---------------------------------------------------------------------------
// 5) lin_w f32 -> bf16, K-padded, row-major [OUT_CH, KP]
// ---------------------------------------------------------------------------
__global__ void prep_b_kernel(const float* __restrict__ linw,
                              __hip_bfloat16* __restrict__ Bbf) {
  int i = blockIdx.x * blockDim.x + threadIdx.x;
  if (i >= OUT_CHv * KPv) return;
  int row = i / KPv, col = i - row * KPv;
  float v = (col < KDIMv) ? linw[(size_t)row * KDIMv + col] : 0.0f;
  Bbf[i] = __float2bfloat16(v);
}

// ---------------------------------------------------------------------------
// TDM issue: load a (2 rows x KC cols) bf16 tile of the [512][KP] B tensor
// into LDS.  D# group0/group1 packed per CDNA5 ISA 8.3/8.4:
//   g0: count=1 | lds_addr | global_addr[56:0] | type=2
//   g1: data_size=1(2B), tensor_dim0=KP, tensor_dim1=512, tile=[KC,2],
//       tensor_dim0_stride=KP
// ---------------------------------------------------------------------------
#if HAVE_TDM
__device__ __forceinline__ void tdm_issue_b(const __hip_bfloat16* Bbf, int n0,
                                            int rowlo, int k0, unsigned int ldsOff) {
  unsigned long long ga =
      (unsigned long long)(size_t)(Bbf + (size_t)(n0 + rowlo) * KPv + k0);
  v4u g0;
  g0[0] = 1u;                                   // count=1, user descriptor
  g0[1] = ldsOff;                               // lds_addr (bytes)
  g0[2] = (unsigned int)ga;                     // global_addr[31:0]
  g0[3] = (unsigned int)(ga >> 32) | (2u << 30);// global_addr[56:32] | type=2
  v8i g1;
  g1[0] = (int)(1u << 16);                                  // data_size=2B
  g1[1] = (int)(((unsigned)KPv & 0xFFFFu) << 16);           // tensor_dim0 lo
  g1[2] = (int)((((unsigned)KPv >> 16) & 0xFFFFu) |
                (((unsigned)OUT_CHv & 0xFFFFu) << 16));     // d0 hi | dim1 lo
  g1[3] = (int)((((unsigned)OUT_CHv >> 16) & 0xFFFFu) |
                ((unsigned)KCv << 16));                     // d1 hi | tile_dim0
  g1[4] = 2;                                                // tile_dim1 = 2 rows
  g1[5] = (int)(unsigned)KPv;                               // dim0_stride lo
  g1[6] = 0;                                                // stride hi | d1s lo
  g1[7] = 0;
  v4i z4 = {0, 0, 0, 0};
#if defined(__clang_major__) && __clang_major__ >= 23
  v8i z8 = {0, 0, 0, 0, 0, 0, 0, 0};
  __builtin_amdgcn_tensor_load_to_lds(g0, g1, z4, z4, z8, 0);
#else
  __builtin_amdgcn_tensor_load_to_lds(g0, g1, z4, z4, 0);
#endif
}
#endif

// ---------------------------------------------------------------------------
// 6) Big GEMM: C[16384,512] = A[16384,4144] * lin_w^T, bf16 WMMA, f32 accum.
//    Block = 8 waves = 128x16 tile sharing one N-tile.  B panel is staged in
//    LDS in KC-chunks via tensor_load_to_lds (double-buffered, TENSORcnt),
//    each wave DMA-ing its own 2 rows; A streams from global (L2-resident).
//    Per-lane operand builds follow the ISA bf16 WMMA VGPR layouts.
// ---------------------------------------------------------------------------
__global__ void __launch_bounds__(256)
gemm_kernel(const __hip_bfloat16* __restrict__ Abf,
            const __hip_bfloat16* __restrict__ Bbf,
            const float* __restrict__ linb, float* __restrict__ out1) {
  __shared__ unsigned short Blds[2][16][KCv];   // 26.6 KB double buffer

  const int tid  = threadIdx.x;
  const int lane = tid & 31;
  const int wv   = __builtin_amdgcn_readfirstlane(tid >> 5);  // scalar wave id
  const int tn   = blockIdx.x & 31;
  const int tb   = blockIdx.x >> 5;
  const int m0   = tb * 128 + wv * 16;
  const int n0   = tn * 16;
  const int r    = lane & 15, half = lane >> 4;

  const uint4* arow = (const uint4*)(Abf + (size_t)(m0 + r) * KPv);

#if HAVE_TDM
  const unsigned int ldsBase = (unsigned int)(size_t)(&Blds[0][0][0]);
  const int rowlo = wv * 2;                 // this wave's 2 B rows
  tdm_issue_b(Bbf, n0, rowlo, 0, ldsBase + (unsigned)(rowlo * KCv) * 2u);
#endif

  v8f acc = {};
  for (int c = 0; c < NCHUNK; ++c) {
    const int k0  = c * KCv;
    const int buf = c & 1;
#if HAVE_TDM
    __builtin_amdgcn_s_wait_tensorcnt(0);   // my 2 rows of chunk c are in LDS
    __syncthreads();                        // everyone's rows are in LDS
    if (c + 1 < NCHUNK)                     // overlap DMA of chunk c+1
      tdm_issue_b(Bbf, n0, rowlo, k0 + KCv,
                  ldsBase + (unsigned)(((buf ^ 1) * 16 + rowlo) * KCv) * 2u);
#else
    // fallback: cooperative copy of chunk c into the buffer
    for (int t = tid; t < 16 * (KCv / 2); t += 256) {
      int row = t / (KCv / 2), col = t - row * (KCv / 2);
      ((unsigned int*)&Blds[buf][row][0])[col] =
          *(const unsigned int*)(Bbf + (size_t)(n0 + row) * KPv + k0 + col * 2);
    }
    __syncthreads();
#endif
    const uint4* bl = (const uint4*)(&Blds[buf][r][0]);   // 52 uint4 per row
    __builtin_prefetch((const void*)(arow + ((k0 + KCv) >> 3)), 0, 1);
#pragma unroll 4
    for (int kk = 0; kk < KCv; kk += 32) {
      const int ug = (k0 + kk) >> 3;        // global uint4 index
      const int ul = kk >> 3;               // LDS uint4 index
      ABu ua, ub;
      ua.u[0] = arow[ug + half];            // A: K = kb + 8*half + [0..7]
      ua.u[1] = arow[ug + half + 2];        // A: K = kb + 16 + 8*half + [0..7]
      ub.u[0] = bl[ul + half];              // B: K = kb + 8*half + [0..7]
      ub.u[1] = bl[ul + half + 2];          // B: K = kb + 16 + 8*half + [0..7]
      acc = __builtin_amdgcn_wmma_f32_16x16x32_bf16(false, ua.v, false, ub.v,
                                                    (short)0, acc, false, false);
    }
    __syncthreads();                        // done reading buf before reuse
  }

  const int n = n0 + r;
  const float bias = linb[n];
#pragma unroll
  for (int j = 0; j < 8; ++j) {             // D VGPR j: M = j + 8*half
    const int m = m0 + j + half * 8;
    float v = acc[j] + bias;
    v = v > 0.0f ? v : 0.1f * v;            // LeakyReLU(0.1)
    const int bb = m >> 11, s = m & (NPOINTv - 1);
    out1[((size_t)bb * OUT_CHv + n) * NPOINTv + s] = v;
  }
}

// ---------------------------------------------------------------------------
extern "C" void kernel_launch(void* const* d_in, const int* in_sizes, int n_in,
                              void* d_out, int out_size, void* d_ws, size_t ws_size,
                              hipStream_t stream) {
  const float* xyz  = (const float*)d_in[0];
  const float* pts  = (const float*)d_in[1];
  const float* w1   = (const float*)d_in[2];
  const float* b1   = (const float*)d_in[3];
  const float* w2   = (const float*)d_in[4];
  const float* b2   = (const float*)d_in[5];
  const float* w3   = (const float*)d_in[6];
  const float* b3   = (const float*)d_in[7];
  const float* linw = (const float*)d_in[8];
  const float* linb = (const float*)d_in[9];

  // d_out: new_xyz [B,3,S] | out [B,512,S] | fps_idx [B,S] (int bits)
  float* out0   = (float*)d_out;
  float* out1   = out0 + (size_t)Bv * 3 * NPOINTv;
  int*   fpsidx = (int*)(out1 + (size_t)Bv * OUT_CHv * NPOINTv);

  // workspace carve-up (~143 MB): A bf16 | B bf16 | new_xyz | knn idx
  char* ws = (char*)d_ws;
  const size_t A_BYTES   = (size_t)Mrows * KPv * sizeof(__hip_bfloat16);
  const size_t B_BYTES   = (size_t)OUT_CHv * KPv * sizeof(__hip_bfloat16);
  const size_t NXZ_BYTES = (size_t)Bv * NPOINTv * 3 * sizeof(float);
  __hip_bfloat16* Abf = (__hip_bfloat16*)ws;
  __hip_bfloat16* Bbf = (__hip_bfloat16*)(ws + A_BYTES);
  float* nxz    = (float*)(ws + A_BYTES + B_BYTES);
  int*   knnidx = (int*)(ws + A_BYTES + B_BYTES + NXZ_BYTES);

  fps_kernel<<<Bv, 1024, 0, stream>>>(xyz, fpsidx);
  gather_newxyz_kernel<<<(Bv * NPOINTv + 255) / 256, 256, 0, stream>>>(xyz, fpsidx, nxz, out0);
  prep_b_kernel<<<(OUT_CHv * KPv + 255) / 256, 256, 0, stream>>>(linw, Bbf);
  knn_kernel<<<Bv * NPOINTv, 32, 0, stream>>>(xyz, nxz, knnidx);
  agg_kernel<<<Bv * NPOINTv, 256, 0, stream>>>(xyz, pts, w1, b1, w2, b2, w3, b3,
                                               nxz, knnidx, Abf);
  gemm_kernel<<<(Mrows / 128) * (OUT_CHv / 16), 256, 0, stream>>>(Abf, Bbf, linb, out1);
}
